// LNNTradingModel_54503134986339
// MI455X (gfx1250) — matrix operand order repaired and checked by reading, not compiled, not run on hardware
//
#include <hip/hip_runtime.h>
#include <math.h>

typedef unsigned short u16;
typedef __attribute__((ext_vector_type(16))) __bf16 v16bf;
typedef __attribute__((ext_vector_type(8)))  float  v8f;
typedef __attribute__((ext_vector_type(4)))  unsigned int u32x4;
typedef __attribute__((ext_vector_type(8)))  int i32x8;
typedef __attribute__((ext_vector_type(4)))  int i32x4;

#if __has_builtin(__builtin_amdgcn_tensor_load_to_lds)
#define USE_TDM 1
#else
#define USE_TDM 0
#endif

// ---------------- problem dims ----------------
constexpr int Bb = 256, Tt = 512, Ii = 64;
constexpr int NWG = 32, TPB = 256;
// padded hidden widths (multiples of 32 so every K-tile sits in one segment)
constexpr int HIW = 320, HCW = 224, HMW = 32;
constexpr int L0_NKT = 12, L0_OT = 20, L0_N = 306, L0_D = 370, L0_FIN = 64,  L0_SEG0 = 64;
constexpr int L1_NKT = 17, L1_OT = 13, L1_N = 204, L1_D = 510, L1_FIN = 306, L1_SEG0 = 320;
constexpr int L2_NKT = 8,  L2_OT = 1,  L2_N = 2,   L2_D = 206, L2_FIN = 204, L2_SEG0 = 224;
// packed-weight block per (layer, otile): 3 matmuls x nkt x (32x16 bf16)
constexpr int L0_BLK = 3 * L0_NKT * 512;
constexpr int L1_BLK = 3 * L1_NKT * 512;
constexpr int L2_BLK = 3 * L2_NKT * 512;
constexpr long long PW_L0 = 0;
constexpr long long PW_L1 = PW_L0 + (long long)L0_OT * L0_BLK;
constexpr long long PW_L2 = PW_L1 + (long long)L1_OT * L1_BLK;
constexpr long long PW_TOT = PW_L2 + L2_BLK;
// LDS slots (u16 units)
constexpr int SA = 0, SBk = L0_BLK, SC = 2 * L0_BLK, SD = SC + L1_BLK;
constexpr int SM_U16 = SD + L2_BLK;            // 75264 u16 = 150528 bytes of LDS

// ---------------- helpers ----------------
__device__ __forceinline__ u16 f2bf(float f) {
  unsigned u = __float_as_uint(f);
  unsigned r = (u + 0x7FFFu + ((u >> 16) & 1u)) >> 16;
  return (u16)r;
}
__device__ __forceinline__ float bf2f(u16 v) {
  return __uint_as_float(((unsigned)v) << 16);
}

union Frag { v16bf v; uint4 q[2]; };

__device__ __forceinline__ void grid_sync(unsigned* bar) {
  __syncthreads();
  if (threadIdx.x == 0) {
    __threadfence();
    unsigned gen = __hip_atomic_load(&bar[1], __ATOMIC_RELAXED, __HIP_MEMORY_SCOPE_AGENT);
    unsigned arr = __hip_atomic_fetch_add(&bar[0], 1u, __ATOMIC_ACQ_REL, __HIP_MEMORY_SCOPE_AGENT);
    if (arr == NWG - 1) {
      __hip_atomic_store(&bar[0], 0u, __ATOMIC_RELAXED, __HIP_MEMORY_SCOPE_AGENT);
      __hip_atomic_fetch_add(&bar[1], 1u, __ATOMIC_RELEASE, __HIP_MEMORY_SCOPE_AGENT);
    } else {
      while (__hip_atomic_load(&bar[1], __ATOMIC_ACQUIRE, __HIP_MEMORY_SCOPE_AGENT) == gen)
        __builtin_amdgcn_s_sleep(2);
    }
  }
  __syncthreads();
}

#if USE_TDM
// One-shot 1-D TDM copy: n bf16 elements from global -> LDS (byte offset).
// D# per CDNA5 ISA: group0 {count=1, lds_addr, global_addr,
// type=2}; group1 {data_size=1(2B), tensor_dim0=n, tensor_dim1=1, tile_dim0=n,
// tensor_dim0_stride=n}. Issued once per wave (TDM ignores EXEC).
__device__ __forceinline__ void tdm_copy(unsigned ldsByteOff, const u16* src, unsigned n) {
  unsigned long long ga = (unsigned long long)src;
  u32x4 g0;
  g0[0] = 1u;                                            // count=1, user mode
  g0[1] = ldsByteOff;                                    // lds_addr
  g0[2] = (unsigned)(ga & 0xFFFFFFFFull);                // global_addr lo
  g0[3] = (unsigned)((ga >> 32) & 0x01FFFFFFull) | (2u << 30); // addr hi | type=2
  i32x8 g1;
  g1[0] = (int)(1u << 16);                               // wg_mask=0, data_size=1 (2B)
  g1[1] = (int)((n & 0xFFFFu) << 16);                    // tensor_dim0[15:0]
  g1[2] = (int)((n >> 16) | (1u << 16));                 // tensor_dim0[31:16] | tensor_dim1=1
  g1[3] = (int)((n & 0xFFFFu) << 16);                    // tensor_dim1 hi=0 | tile_dim0=n
  g1[4] = 0;                                             // tile_dim1=0, tile_dim2=0 (unused)
  g1[5] = (int)n;                                        // tensor_dim0_stride lo
  g1[6] = 0;
  g1[7] = 0;
  i32x4 z = {0, 0, 0, 0};
#if defined(__clang_major__) && (__clang_major__ >= 23)
  i32x8 z8 = {0, 0, 0, 0, 0, 0, 0, 0};
  __builtin_amdgcn_tensor_load_to_lds(g0, g1, z, z, z8, 0);
#else
  __builtin_amdgcn_tensor_load_to_lds(g0, g1, z, z, 0);
#endif
}
#endif

// One 16x16 output tile of one cell: 3 GEMMs over nkt K-tiles + liquid-cell
// nonlinearity, writing bf16 h back to the global state buffer.
__device__ __forceinline__ void cell_tile(
    int lane,
    const u16* __restrict__ a0, int a0stride, int a0cols,
    const u16* __restrict__ a1, int a1stride,
    const u16* __restrict__ wlds, int nkt,
    int rtile, int ocol,
    const float* __restrict__ bf1, const float* __restrict__ bf2,
    const float* __restrict__ bta, const float* __restrict__ btb, int nvalid,
    u16* __restrict__ hout, int ostride)
{
  v8f acc1 = {}, acc2 = {}, acc3 = {};
  const int m    = lane & 15;
  const int hi16 = (lane & 16) ? 1 : 0;
  const int arow = rtile * 16 + m;
  for (int kt = 0; kt < nkt; ++kt) {
    const int cbase = kt * 32;
    const u16* seg;
    if (cbase < a0cols) seg = a0 + arow * a0stride + cbase;
    else                seg = a1 + arow * a1stride + (cbase - a0cols);
    const u16* ap = seg + (hi16 ? 8 : 0);           // A layout: lane<16 K{0..7,16..23}
    Frag A;
    A.q[0] = *(const uint4*)ap;
    A.q[1] = *(const uint4*)(ap + 16);
    const u16* wp = wlds + kt * 512 + lane * 16;    // fragment-ready B blocks
    Frag B1, B2, B3;
    B1.q[0] = *(const uint4*)wp;                   B1.q[1] = *(const uint4*)(wp + 8);
    B2.q[0] = *(const uint4*)(wp + nkt * 512);     B2.q[1] = *(const uint4*)(wp + nkt * 512 + 8);
    B3.q[0] = *(const uint4*)(wp + 2 * nkt * 512); B3.q[1] = *(const uint4*)(wp + 2 * nkt * 512 + 8);
    acc1 = __builtin_amdgcn_wmma_f32_16x16x32_bf16(false, A.v, false, B1.v, (short)0, acc1, false, false);
    acc2 = __builtin_amdgcn_wmma_f32_16x16x32_bf16(false, A.v, false, B2.v, (short)0, acc2, false, false);
    acc3 = __builtin_amdgcn_wmma_f32_16x16x32_bf16(false, A.v, false, B3.v, (short)0, acc3, false, false);
  }
  const int col = ocol + m;
  const float b1 = (col < nvalid) ? bf1[col] : 0.f;
  const float b2 = (col < nvalid) ? bf2[col] : 0.f;
  const float bd = (col < nvalid) ? (btb[col] - bta[col]) : 0.f;
#pragma unroll
  for (int r = 0; r < 8; ++r) {
    const int row = rtile * 16 + (hi16 ? 8 : 0) + r;
    const float f1 = tanhf(acc1[r] + b1);
    const float f2 = tanhf(acc2[r] + b2);
    const float tt = 0.5f * tanhf(0.5f * (acc3[r] + bd)) + 0.5f;  // sigmoid via HW tanh
    hout[row * ostride + col] = f2bf(f1 + tt * (f2 - f1));
  }
}

// ---------------- prep kernels ----------------
__global__ void k_init(unsigned* bar) { bar[0] = 0u; bar[1] = 0u; }

__global__ void k_xcast(const float* __restrict__ x, u16* __restrict__ xb, int n) {
  int i = blockIdx.x * blockDim.x + threadIdx.x;
  if (i < n) xb[i] = f2bf(x[i]);
}

__global__ void k_hinit(const float* __restrict__ h0,
                        u16* HI0, u16* HI1, u16* HC0, u16* HC1, u16* HM0, u16* HM1) {
  int i = blockIdx.x * blockDim.x + threadIdx.x;
  if (i < Bb * HIW) {
    int b = i / HIW, u = i - b * HIW;
    HI0[i] = (u < L0_N) ? f2bf(h0[b * 512 + u]) : (u16)0;
    HI1[i] = 0;
  }
  if (i < Bb * HCW) {
    int b = i / HCW, u = i - b * HCW;
    HC0[i] = (u < L1_N) ? f2bf(h0[b * 512 + 306 + u]) : (u16)0;
    HC1[i] = 0;
  }
  if (i < Bb * HMW) {
    int b = i / HMW, u = i - b * HMW;
    HM0[i] = (u < L2_N) ? f2bf(h0[b * 512 + 510 + u]) : (u16)0;
    HM1[i] = 0;
  }
}

// Pack masked/folded weights into fragment-ready bf16 B-blocks.
__global__ void k_pack(const float* __restrict__ w1, const float* __restrict__ w2,
                       const float* __restrict__ taw, const float* __restrict__ tbw,
                       const int* __restrict__ mask, u16* __restrict__ dst,
                       int N, int D, int FIN, int SEG0, int NKT, int OT) {
  long long total = (long long)OT * 3 * NKT * 512;
  long long idx = (long long)blockIdx.x * blockDim.x + threadIdx.x;
  if (idx >= total) return;
  int j = (int)(idx & 15);
  int lane = (int)((idx >> 4) & 31);
  long long rest = idx >> 9;
  int kt = (int)(rest % NKT);
  long long q = rest / NKT;
  int mm = (int)(q % 3);
  int o = (int)(q / 3);
  int nl = lane & 15;
  int kl = (j < 8) ? ((lane >> 4) * 8 + j) : (16 + (lane >> 4) * 8 + (j - 8));
  int K = kt * 32 + kl;
  int Ncol = o * 16 + nl;
  float v = 0.f;
  if (Ncol < N) {
    int c = -1;
    if (K < SEG0) { if (K < FIN) c = K; }
    else { int h = K - SEG0; if (h < N) c = FIN + h; }
    if (c >= 0) {
      if (mm == 0)      v = w1[(long long)Ncol * D + c] * (float)mask[(long long)Ncol * D + c];
      else if (mm == 1) v = w2[(long long)Ncol * D + c] * (float)mask[(long long)Ncol * D + c];
      else              v = tbw[(long long)Ncol * D + c] - taw[(long long)Ncol * D + c];
    }
  }
  dst[idx] = f2bf(v);
}

// ---------------- persistent main kernel ----------------
struct Pm {
  const u16* xb; const u16* packW;
  u16 *HI0, *HI1, *HC0, *HC1, *HM0, *HM1;
  const float *b1_0, *b2_0, *ta_0, *tb_0;
  const float *b1_1, *b2_1, *ta_1, *tb_1;
  const float *b1_2, *b2_2, *ta_2, *tb_2;
  const float *fcw, *fcb;
  float* out;
  unsigned* bar;
};

__global__ void __launch_bounds__(TPB, 1) k_main(Pm p) {
  extern __shared__ u16 sm[];
  const int wg = blockIdx.x, tid = threadIdx.x;
  const int wave = tid >> 5, lane = tid & 31;
  // work mapping (NWG=32, 8 waves/WG = 256 wave-workers):
  const int po = wg >> 1;                 // L0 primary otile (0..15)
  const int rp = (wg & 1) * 8 + wave;     // its rtile for this wave
  const int eo = 16 + (wg >> 3);          // L0 extra otile (16..19), waves 0..1
  const int re = (wg & 7) * 2;            // extra rtile base for this WG
  const bool hasL1 = (wg < 26);
  const bool hasL2 = (wg < 16);           // L2 rtile = wg, on wave 7

  // --- stage this WGP's weight slices into LDS once (resident for all steps)
#if USE_TDM
  if (wave == 0) {
    tdm_copy((unsigned)(SA * 2), p.packW + PW_L0 + (long long)po * L0_BLK, L0_BLK);
    tdm_copy((unsigned)(SBk * 2), p.packW + PW_L0 + (long long)eo * L0_BLK, L0_BLK);
    if (hasL1) tdm_copy((unsigned)(SC * 2), p.packW + PW_L1 + (long long)po * L1_BLK, L1_BLK);
    if (hasL2) tdm_copy((unsigned)(SD * 2), p.packW + PW_L2, L2_BLK);
    __builtin_amdgcn_s_wait_tensorcnt(0);
  }
#else
  {
    auto cp = [&](int dstOff, long long srcOff, int n) {
      const uint4* s = (const uint4*)(p.packW + srcOff);
      uint4* d = (uint4*)(sm + dstOff);
      for (int i = tid; i < (n >> 3); i += TPB) d[i] = s[i];
    };
    cp(SA, PW_L0 + (long long)po * L0_BLK, L0_BLK);
    cp(SBk, PW_L0 + (long long)eo * L0_BLK, L0_BLK);
    if (hasL1) cp(SC, PW_L1 + (long long)po * L1_BLK, L1_BLK);
    if (hasL2) cp(SD, PW_L2, L2_BLK);
  }
#endif
  __syncthreads();

  for (int t = 0; t < Tt; ++t) {
    u16* HIin  = (t & 1) ? p.HI1 : p.HI0;
    u16* HIout = (t & 1) ? p.HI0 : p.HI1;
    u16* HCin  = (t & 1) ? p.HC1 : p.HC0;
    u16* HCout = (t & 1) ? p.HC0 : p.HC1;

    // phase A: deferred layer2(t-1) (hm only feeds layer2) + layer0(t)
    if (hasL2 && wave == 7 && t > 0) {
      int tp = t - 1;
      u16* HCoP = (tp & 1) ? p.HC0 : p.HC1;   // hc_out of step t-1
      u16* HMi  = (tp & 1) ? p.HM1 : p.HM0;
      u16* HMo  = (tp & 1) ? p.HM0 : p.HM1;
      cell_tile(lane, HCoP, HCW, L2_SEG0, HMi, HMW, sm + SD, L2_NKT, wg, 0,
                p.b1_2, p.b2_2, p.ta_2, p.tb_2, L2_N, HMo, HMW);
    }
    const u16* xt = p.xb + t * Ii;             // A rows stride Tt*Ii
    cell_tile(lane, xt, Tt * Ii, L0_SEG0, HIin, HIW, sm + SA, L0_NKT, rp, po * 16,
              p.b1_0, p.b2_0, p.ta_0, p.tb_0, L0_N, HIout, HIW);
    if (wave < 2) {                            // extra L0 otiles 16..19
      cell_tile(lane, xt, Tt * Ii, L0_SEG0, HIin, HIW, sm + SBk, L0_NKT, re + wave, eo * 16,
                p.b1_0, p.b2_0, p.ta_0, p.tb_0, L0_N, HIout, HIW);
    }
    grid_sync(p.bar);

    // phase B: layer1(t) reads fresh hi + previous hc
    if (hasL1) {
      cell_tile(lane, HIout, HIW, L1_SEG0, HCin, HCW, sm + SC, L1_NKT, rp, po * 16,
                p.b1_1, p.b2_1, p.ta_1, p.tb_1, L1_N, HCout, HCW);
    }
    grid_sync(p.bar);
  }

  // final layer2 (t = 511)
  if (hasL2 && wave == 7) {
    int tp = Tt - 1;
    u16* HCoP = (tp & 1) ? p.HC0 : p.HC1;
    u16* HMi  = (tp & 1) ? p.HM1 : p.HM0;
    u16* HMo  = (tp & 1) ? p.HM0 : p.HM1;
    cell_tile(lane, HCoP, HCW, L2_SEG0, HMi, HMW, sm + SD, L2_NKT, wg, 0,
              p.b1_2, p.b2_2, p.ta_2, p.tb_2, L2_N, HMo, HMW);
  }
  grid_sync(p.bar);

  // epilogue: predictions = hm @ fc_w.T + fc_b, then h = concat(hi,hc,hm) in f32
  const int g = wg * TPB + tid;
  if (g < Bb * 2) {
    int b = g >> 1, jj = g & 1;
    float h0v = bf2f(p.HM0[b * HMW + 0]);
    float h1v = bf2f(p.HM0[b * HMW + 1]);
    p.out[g] = h0v * p.fcw[jj * 2 + 0] + h1v * p.fcw[jj * 2 + 1] + p.fcb[jj];
  }
  for (int idx = g; idx < Bb * 512; idx += NWG * TPB) {
    int b = idx >> 9, u = idx & 511;
    float v;
    if (u < 306)      v = bf2f(p.HI0[b * HIW + u]);
    else if (u < 510) v = bf2f(p.HC0[b * HCW + (u - 306)]);
    else              v = bf2f(p.HM0[b * HMW + (u - 510)]);
    p.out[512 + idx] = v;
  }
}

// ---------------- host ----------------
extern "C" void kernel_launch(void* const* d_in, const int* in_sizes, int n_in,
                              void* d_out, int out_size, void* d_ws, size_t ws_size,
                              hipStream_t stream) {
  const float* x  = (const float*)d_in[0];
  const float* h0 = (const float*)d_in[1];
  const float* W[3][4]; const float* Bv[3][4]; const int* Mk[3];
  for (int l = 0; l < 3; ++l) {
    int base = 2 + l * 9;
    W[l][0] = (const float*)d_in[base + 0]; Bv[l][0] = (const float*)d_in[base + 1]; // ff1
    W[l][1] = (const float*)d_in[base + 2]; Bv[l][1] = (const float*)d_in[base + 3]; // ff2
    W[l][2] = (const float*)d_in[base + 4]; Bv[l][2] = (const float*)d_in[base + 5]; // ta
    W[l][3] = (const float*)d_in[base + 6]; Bv[l][3] = (const float*)d_in[base + 7]; // tb
    Mk[l]   = (const int*)d_in[base + 8];
  }
  const float* fcw = (const float*)d_in[29];
  const float* fcb = (const float*)d_in[30];
  float* out = (float*)d_out;

  // workspace carve-out (256B aligned regions)
  char* ws = (char*)d_ws;
  size_t off = 0;
  auto take = [&](size_t bytes) -> char* {
    char* q = ws + off;
    off = (off + bytes + 255) & ~(size_t)255;
    return q;
  };
  unsigned* bar = (unsigned*)take(2 * sizeof(unsigned));
  u16* xb  = (u16*)take((size_t)Bb * Tt * Ii * 2);
  u16* HI0 = (u16*)take((size_t)Bb * HIW * 2);
  u16* HI1 = (u16*)take((size_t)Bb * HIW * 2);
  u16* HC0 = (u16*)take((size_t)Bb * HCW * 2);
  u16* HC1 = (u16*)take((size_t)Bb * HCW * 2);
  u16* HM0 = (u16*)take((size_t)Bb * HMW * 2);
  u16* HM1 = (u16*)take((size_t)Bb * HMW * 2);
  u16* packW = (u16*)take((size_t)PW_TOT * 2);

  k_init<<<1, 1, 0, stream>>>(bar);
  {
    int n = Bb * Tt * Ii;
    k_xcast<<<(n + TPB - 1) / TPB, TPB, 0, stream>>>(x, xb, n);
  }
  k_hinit<<<(Bb * HIW + TPB - 1) / TPB, TPB, 0, stream>>>(h0, HI0, HI1, HC0, HC1, HM0, HM1);

  {
    long long tot0 = (long long)L0_OT * 3 * L0_NKT * 512;
    k_pack<<<(unsigned)((tot0 + TPB - 1) / TPB), TPB, 0, stream>>>(
        W[0][0], W[0][1], W[0][2], W[0][3], Mk[0], packW + PW_L0,
        L0_N, L0_D, L0_FIN, L0_SEG0, L0_NKT, L0_OT);
    long long tot1 = (long long)L1_OT * 3 * L1_NKT * 512;
    k_pack<<<(unsigned)((tot1 + TPB - 1) / TPB), TPB, 0, stream>>>(
        W[1][0], W[1][1], W[1][2], W[1][3], Mk[1], packW + PW_L1,
        L1_N, L1_D, L1_FIN, L1_SEG0, L1_NKT, L1_OT);
    long long tot2 = (long long)L2_OT * 3 * L2_NKT * 512;
    k_pack<<<(unsigned)((tot2 + TPB - 1) / TPB), TPB, 0, stream>>>(
        W[2][0], W[2][1], W[2][2], W[2][3], Mk[2], packW + PW_L2,
        L2_N, L2_D, L2_FIN, L2_SEG0, L2_NKT, L2_OT);
  }

  Pm p;
  p.xb = xb; p.packW = packW;
  p.HI0 = HI0; p.HI1 = HI1; p.HC0 = HC0; p.HC1 = HC1; p.HM0 = HM0; p.HM1 = HM1;
  p.b1_0 = Bv[0][0]; p.b2_0 = Bv[0][1]; p.ta_0 = Bv[0][2]; p.tb_0 = Bv[0][3];
  p.b1_1 = Bv[1][0]; p.b2_1 = Bv[1][1]; p.ta_1 = Bv[1][2]; p.tb_1 = Bv[1][3];
  p.b1_2 = Bv[2][0]; p.b2_2 = Bv[2][1]; p.ta_2 = Bv[2][2]; p.tb_2 = Bv[2][3];
  p.fcw = fcw; p.fcb = fcb; p.out = out; p.bar = bar;

  k_main<<<NWG, TPB, SM_U16 * 2, stream>>>(p);
}
